// NnBoard768_13580686590391
// MI455X (gfx1250) — compile-verified, matrix operand order checked by reading, not run on hardware
//
#include <hip/hip_runtime.h>
#include <math.h>

// ---- problem constants (from reference) ----
#define FT_IN   40960
#define FT_OUT  512
#define BATCH   8192
#define NNZ     32
#define HID     (2 * FT_OUT)   // 1024

typedef _Float16 f16;
typedef __attribute__((ext_vector_type(4)))  _Float16 v4h;
typedef __attribute__((ext_vector_type(8)))  _Float16 v8h;
typedef __attribute__((ext_vector_type(16))) _Float16 v16h;
typedef __attribute__((ext_vector_type(4)))  float    v4f;
typedef __attribute__((ext_vector_type(8)))  float    v8f;

static __device__ inline v16h combine8(v8h lo, v8h hi) {
    v16h r;
#pragma unroll
    for (int e = 0; e < 8; ++e) { r[e] = lo[e]; r[e + 8] = hi[e]; }
    return r;
}

// ---------------------------------------------------------------------------
// Kernel A: feature transform (sparse gather-accumulate), one wave32 per
// (position, side). 512-float accumulator = 4 x float4 per lane.
// Gathers are 2KB rows -> 32 lanes x float4 x 4 chunks, fully coalesced
// global_load_b128. Indices/values broadcast via wave shuffles; next gather
// row is prefetched (global_prefetch_b8) one step ahead.
// Output: clipped hidden stored as f16 [BATCH, 1024] in workspace.
// ---------------------------------------------------------------------------
__global__ __launch_bounds__(256) void nnue_ft_kernel(
    const int*   __restrict__ stm,
    const int*   __restrict__ nstm,
    const float* __restrict__ values,
    const float* __restrict__ Wft,
    const float* __restrict__ bft,
    f16*         __restrict__ hidden)
{
    const int lane = threadIdx.x & 31;
    const int wave = (blockIdx.x * blockDim.x + threadIdx.x) >> 5; // global wave
    const int b    = wave >> 1;
    const int side = wave & 1;

    const int*   idxrow = (side ? nstm : stm) + b * NNZ;
    const float* valrow = values + b * NNZ;

    // cooperative load: lane k owns entry k of this position's sparse row
    const int   myidx = idxrow[lane];
    const float myval = valrow[lane];

    // accumulators seeded with bias (each side adds b_ft once, per reference)
    v4f acc0 = *(const v4f*)(bft + 0 * 128 + lane * 4);
    v4f acc1 = *(const v4f*)(bft + 1 * 128 + lane * 4);
    v4f acc2 = *(const v4f*)(bft + 2 * 128 + lane * 4);
    v4f acc3 = *(const v4f*)(bft + 3 * 128 + lane * 4);

    for (int k = 0; k < NNZ; ++k) {
        const int   idx = __shfl(myidx, k, 32);
        const float v   = __shfl(myval, k, 32);
        const float* row = Wft + (size_t)idx * FT_OUT;

        // prefetch next gather row unconditionally (wraps to row 0 on last
        // iter -- harmless): 32 lanes x 64B spacing covers the 2KB row
        const int nidx = __shfl(myidx, (k + 1) & (NNZ - 1), 32);
        __builtin_prefetch(Wft + (size_t)nidx * FT_OUT + lane * 16, 0, 0);

        const v4f r0 = *(const v4f*)(row + 0 * 128 + lane * 4);
        const v4f r1 = *(const v4f*)(row + 1 * 128 + lane * 4);
        const v4f r2 = *(const v4f*)(row + 2 * 128 + lane * 4);
        const v4f r3 = *(const v4f*)(row + 3 * 128 + lane * 4);
        acc0 += v * r0;
        acc1 += v * r1;
        acc2 += v * r2;
        acc3 += v * r3;
    }

    // clip [0,1], convert to f16, store 4 x (4 halves = 8B) per lane
    f16* base = hidden + (size_t)b * HID + side * FT_OUT;
    v4f accs[4] = {acc0, acc1, acc2, acc3};
#pragma unroll
    for (int j = 0; j < 4; ++j) {
        v4h h;
#pragma unroll
        for (int i = 0; i < 4; ++i) {
            float x = accs[j][i];
            x = fminf(fmaxf(x, 0.0f), 1.0f);
            h[i] = (f16)x;
        }
        *(v4h*)(base + j * 128 + lane * 4) = h;
    }
}

// ---------------------------------------------------------------------------
// Kernel P: one-time convert W_out (1024 f32) to f16 in workspace so the
// WMMA loop loads v8h directly (no per-iteration cvt chains).
// ---------------------------------------------------------------------------
__global__ __launch_bounds__(256) void nnue_wout_prep_kernel(
    const float* __restrict__ Wout, f16* __restrict__ WoutH)
{
    const int i = blockIdx.x * blockDim.x + threadIdx.x;
    if (i < HID) WoutH[i] = (f16)Wout[i];
}

// ---------------------------------------------------------------------------
// Kernel B: output layer as WMMA GEMV. One wave32 computes 16 positions:
//   D[16x16] += A[16x32 f16 hidden tile] x B[32x16 f16].
// Trick: B is built with ALL columns equal to the W_out chunk, so every
// column of C holds the same dot product -- no lane masking, no divergence,
// B setup = two uniform 16B loads. K=1024 -> 32 WMMAs per wave, split over
// two independent accumulators (c0/c1) to break the D->C serial chain.
// Lanes 0 and 16 hold C column data (M=0..7 / M=8..15): bias+sigmoid+store.
// ---------------------------------------------------------------------------
__global__ __launch_bounds__(256) void nnue_out_kernel(
    const f16*   __restrict__ hidden,
    const f16*   __restrict__ WoutH,
    const float* __restrict__ bout,
    float*       __restrict__ out)
{
    const int lane = threadIdx.x & 31;
    const int wave = (blockIdx.x * blockDim.x + threadIdx.x) >> 5;
    const int m0   = wave * 16;

    const int half = lane >> 4;   // 0: K 0..7/16..23, 1: K 8..15/24..31
    const int mn   = lane & 15;   // M index for A (N index for B)
    const int kb   = half * 8;

    const f16* rowA = hidden + (size_t)(m0 + mn) * HID + kb;
    const f16* rowB = WoutH + kb;

    v8f c0 = {};
    v8f c1 = {};
    for (int kk = 0; kk < HID; kk += 64) {
        // chunk 0: K = kk .. kk+31
        const v16h a0 = combine8(*(const v8h*)(rowA + kk),
                                 *(const v8h*)(rowA + kk + 16));
        const v16h b0 = combine8(*(const v8h*)(rowB + kk),
                                 *(const v8h*)(rowB + kk + 16));
        c0 = __builtin_amdgcn_wmma_f32_16x16x32_f16(
                 false, a0, false, b0, (short)0, c0, false, false);

        // chunk 1: K = kk+32 .. kk+63 (independent accumulator)
        const v16h a1 = combine8(*(const v8h*)(rowA + kk + 32),
                                 *(const v8h*)(rowA + kk + 48));
        const v16h b1 = combine8(*(const v8h*)(rowB + kk + 32),
                                 *(const v8h*)(rowB + kk + 48));
        c1 = __builtin_amdgcn_wmma_f32_16x16x32_f16(
                 false, a1, false, b1, (short)0, c1, false, false);
    }
    const v8f c = c0 + c1;

    const float bo = bout[0];
    if (mn == 0) {
        // C layout: lane 0 holds M=0..7 in c[0..7]; lane 16 holds M=8..15
        const int mbase = m0 + half * 8;
#pragma unroll
        for (int r = 0; r < 8; ++r) {
            const float x = c[r] + bo;
            out[mbase + r] = 1.0f / (1.0f + __expf(-x));
        }
    }
}

// ---------------------------------------------------------------------------
extern "C" void kernel_launch(void* const* d_in, const int* in_sizes, int n_in,
                              void* d_out, int out_size, void* d_ws, size_t ws_size,
                              hipStream_t stream) {
    (void)in_sizes; (void)n_in; (void)out_size; (void)ws_size;

    const int*   stm    = (const int*)  d_in[0];  // [B, NNZ] int32
    const int*   nstm   = (const int*)  d_in[1];  // [B, NNZ] int32
    const float* values = (const float*)d_in[2];  // [B, NNZ] f32
    const float* Wft    = (const float*)d_in[3];  // [FT_IN, FT_OUT] f32
    const float* bft    = (const float*)d_in[4];  // [FT_OUT] f32
    const float* Wout   = (const float*)d_in[5];  // [2*FT_OUT, 1] f32
    const float* bout   = (const float*)d_in[6];  // [1] f32

    f16*   hidden = (f16*)d_ws;                       // [B, 1024] f16 (16 MB)
    f16*   WoutH  = (f16*)d_ws + (size_t)BATCH * HID; // [1024] f16 (2 KB)
    float* out    = (float*)d_out;                    // [B, 1] f32

    // Kernel P: convert W_out to f16 (runs concurrently-safe, tiny)
    nnue_wout_prep_kernel<<<(HID + 255) / 256, 256, 0, stream>>>(Wout, WoutH);

    // Kernel A: (BATCH*2) waves, 8 waves (256 threads) per block
    const int wavesA  = BATCH * 2;
    const int blocksA = wavesA / 8;      // 2048
    nnue_ft_kernel<<<blocksA, 256, 0, stream>>>(stm, nstm, values, Wft, bft, hidden);

    // Kernel B: BATCH/16 waves, 8 per block
    const int wavesB  = BATCH / 16;      // 512
    const int blocksB = wavesB / 8;      // 64
    nnue_out_kernel<<<blocksB, 256, 0, stream>>>(hidden, WoutH, bout, out);
}
